// AttentionBlock_55611236549240
// MI455X (gfx1250) — matrix operand order repaired and checked by reading, not compiled, not run on hardware
//
#include <hip/hip_runtime.h>
#include <hip/hip_bf16.h>
#include <stdint.h>

// MI455X / gfx1250: wave32, WMMA bf16 16x16x32.
typedef __attribute__((ext_vector_type(16))) __bf16 v16bf;
typedef __attribute__((ext_vector_type(8)))  float  v8f;

union FragU  { uint4 u[2]; v16bf v; };
union U16x16 { uint4 u[2]; unsigned short s[16]; };

__device__ inline unsigned short f2bf(float x) {
  unsigned int u = __float_as_uint(x);
  u = (u + 0x7FFFu + ((u >> 16) & 1u)) >> 16;   // round-to-nearest-even
  return (unsigned short)u;
}

// Build a v16bf A/B fragment from two 16-byte LDS/global chunks.
__device__ inline v16bf load_frag(const unsigned short* p0, const unsigned short* p1) {
  FragU f;
  f.u[0] = *(const uint4*)p0;
  f.u[1] = *(const uint4*)p1;
  return f.v;
}

#define WMMA_BF16(a, b, c) \
  __builtin_amdgcn_wmma_f32_16x16x32_bf16(false, (a), false, (b), (short)0, (c), false, false)

// ---------------------------------------------------------------- prep: f32 -> bf16 weights
__global__ __launch_bounds__(256) void prep_kernel(const float* __restrict__ wq,
                                                   const float* __restrict__ wp,
                                                   unsigned short* __restrict__ wq_bf,
                                                   unsigned short* __restrict__ wp_bf) {
  int i = blockIdx.x * 256 + threadIdx.x;
  const int n1 = 512 * 1536;
  const int n2 = 512 * 512;
  if (i < n1) wq_bf[i] = f2bf(wq[i]);
  if (i < n2) wp_bf[i] = f2bf(wp[i]);
}

// ---------------------------------------------------------------- QKV GEMM: [16384,512] x [512,1536]
// WG tile: 128 s-rows x 64 n-cols; wave tile 32x64 (2 A frags x 4 B frags = 8 WMMA / k-chunk).
__global__ __launch_bounds__(128) void qkv_kernel(const float* __restrict__ x,
                                                  const unsigned short* __restrict__ wq,
                                                  const float* __restrict__ bq,
                                                  unsigned short* __restrict__ Q,
                                                  unsigned short* __restrict__ K,
                                                  unsigned short* __restrict__ V) {
  __shared__ unsigned short ld_a[128][32];  // [s_local][k_local]
  __shared__ unsigned short ld_b[64][32];   // [n_local][k_local]
  const int n0 = blockIdx.x * 64;
  const int s0 = blockIdx.y * 128;
  const int b  = blockIdx.z;
  const int t    = threadIdx.x;
  const int wave = t >> 5, lane = t & 31, half = lane >> 4, ln = lane & 15;

  const v8f zero8 = {0.f, 0.f, 0.f, 0.f, 0.f, 0.f, 0.f, 0.f};
  v8f acc[2][4] = {{zero8, zero8, zero8, zero8}, {zero8, zero8, zero8, zero8}};

  for (int k0 = 0; k0 < 512; k0 += 32) {
    __syncthreads();
    {
      int kk = t >> 2, sb = (t & 3) * 32;
      const float* xp = x + ((size_t)(b * 512 + k0 + kk)) * 1024 + s0 + sb;
#pragma unroll
      for (int i = 0; i < 8; ++i) {
        float4 f = *(const float4*)(xp + i * 4);
        ld_a[sb + i * 4 + 0][kk] = f2bf(f.x);
        ld_a[sb + i * 4 + 1][kk] = f2bf(f.y);
        ld_a[sb + i * 4 + 2][kk] = f2bf(f.z);
        ld_a[sb + i * 4 + 3][kk] = f2bf(f.w);
      }
      int nb = (t & 3) * 16;
      const unsigned short* wpp = wq + (size_t)(k0 + kk) * 1536 + n0 + nb;
      U16x16 wv;
      wv.u[0] = *(const uint4*)wpp;
      wv.u[1] = *(const uint4*)(wpp + 8);
#pragma unroll
      for (int i = 0; i < 16; ++i) ld_b[nb + i][kk] = wv.s[i];
    }
    __syncthreads();

    const int ab = half ? 8 : 0;                 // A frag K pattern per 16-lane half
    const unsigned short* pa0 = &ld_a[wave * 32 + ln][0];
    const unsigned short* pa1 = &ld_a[wave * 32 + 16 + ln][0];
    v16bf a0 = load_frag(pa0 + ab, pa0 + ab + 16);
    v16bf a1 = load_frag(pa1 + ab, pa1 + ab + 16);
#pragma unroll
    for (int j = 0; j < 4; ++j) {
      const unsigned short* pb = &ld_b[j * 16 + ln][0];
      v16bf bb = load_frag(pb + half * 8, pb + half * 8 + 16);
      acc[0][j] = WMMA_BF16(a0, bb, acc[0][j]);
      acc[1][j] = WMMA_BF16(a1, bb, acc[1][j]);
    }
  }

#pragma unroll
  for (int j = 0; j < 4; ++j) {
    const int nbase = n0 + j * 16;
    const int head = nbase / 192;
    const int rem  = nbase % 192;
    const int type = rem / 64;
    const int d0   = rem % 64;
    unsigned short* dst = (type == 0) ? Q : (type == 1) ? K : V;
    const float bias = bq[nbase + ln];
    const size_t base = ((size_t)(b * 8 + head) * 1024) * 64 + (size_t)(d0 + ln);
#pragma unroll
    for (int m2 = 0; m2 < 2; ++m2) {
#pragma unroll
      for (int r = 0; r < 8; ++r) {
        const int s = s0 + wave * 32 + m2 * 16 + r + half * 8;
        dst[base + (size_t)s * 64] = f2bf(acc[m2][j][r] + bias);
      }
    }
  }
}

// ---------------------------------------------------------------- flash attention, one wave = 16 queries
__global__ __launch_bounds__(128) void attn_kernel(const unsigned short* __restrict__ Q,
                                                   const unsigned short* __restrict__ K,
                                                   const unsigned short* __restrict__ V,
                                                   unsigned short* __restrict__ O) {
  __shared__ unsigned short ld_k[32][64];       // [key_local][d]
  __shared__ unsigned short ld_v[64][32];       // [d][key_local] (transposed)
  __shared__ unsigned short ld_p[4][16][32];    // per-wave P scratch (C->A relayout)
  const int qt = blockIdx.x;                    // query tile of 64 (0..15)
  const int bh = blockIdx.y;                    // b*8+h (0..127)
  const int t    = threadIdx.x;
  const int wave = t >> 5, lane = t & 31, half = lane >> 4, ln = lane & 15;
  const int srow = qt * 64 + wave * 16 + ln;

  // Q A-fragments for both d_k chunks (K=0..31, 32..63), resident in VGPRs all along.
  const unsigned short* qp = Q + ((size_t)bh * 1024 + srow) * 64;
  v16bf qa[2];
#pragma unroll
  for (int c2 = 0; c2 < 2; ++c2) {
    const int b1 = c2 * 32 + (half ? 16 : 0);
    qa[c2] = load_frag(qp + b1, qp + b1 + 8);
  }

  const v8f zero8 = {0.f, 0.f, 0.f, 0.f, 0.f, 0.f, 0.f, 0.f};
  v8f acc[4] = {zero8, zero8, zero8, zero8};
  float mrow[8], lrow[8];
#pragma unroll
  for (int r = 0; r < 8; ++r) { mrow[r] = -3.0e38f; lrow[r] = 0.f; }

  const int ldkey = t >> 2, lddb = (t & 3) * 16;
  const size_t kvrow = (size_t)bh * 1024 + ldkey;

  for (int kc = 0; kc < 32; ++kc) {
    const int key0 = kc * 32;
    __syncthreads();
    {   // cooperative K / V(T) staging: 32 keys x 64 d each
      const unsigned short* kp = K + (kvrow + key0) * 64 + lddb;
      *(uint4*)&ld_k[ldkey][lddb]     = *(const uint4*)kp;
      *(uint4*)&ld_k[ldkey][lddb + 8] = *(const uint4*)(kp + 8);
      const unsigned short* vp = V + (kvrow + key0) * 64 + lddb;
      U16x16 vv;
      vv.u[0] = *(const uint4*)vp;
      vv.u[1] = *(const uint4*)(vp + 8);
#pragma unroll
      for (int i = 0; i < 16; ++i) ld_v[lddb + i][ldkey] = vv.s[i];
    }
    __syncthreads();

    // prefetch next chunk's K/V rows into cache (global_prefetch_b8)
    if (kc + 1 < 32) {
      __builtin_prefetch(K + (kvrow + key0 + 32) * 64 + lddb, 0, 3);
      __builtin_prefetch(V + (kvrow + key0 + 32) * 64 + lddb, 0, 3);
    }

    // scores: two 16-key tiles, K-dim = d_k (2 WMMAs each)
    v8f s0 = zero8, s1 = zero8;
#pragma unroll
    for (int c2 = 0; c2 < 2; ++c2) {
      const unsigned short* pk0 = &ld_k[ln][c2 * 32 + half * 16];
      v16bf kb0 = load_frag(pk0, pk0 + 8);
      s0 = WMMA_BF16(qa[c2], kb0, s0);
      const unsigned short* pk1 = &ld_k[16 + ln][c2 * 32 + half * 16];
      v16bf kb1 = load_frag(pk1, pk1 + 8);
      s1 = WMMA_BF16(qa[c2], kb1, s1);
    }

    // streaming softmax per row; N dim is striped over 16-lane halves -> xor-shuffle reduce
#pragma unroll
    for (int r = 0; r < 8; ++r) {
      float a0 = s0[r] * 0.125f;     // d_k^-0.5 = 1/8
      float a1 = s1[r] * 0.125f;
      float mx = fmaxf(a0, a1);
#pragma unroll
      for (int off = 8; off >= 1; off >>= 1) mx = fmaxf(mx, __shfl_xor(mx, off, 32));
      const float mnew  = fmaxf(mrow[r], mx);
      const float alpha = __expf(mrow[r] - mnew);
      const float p0 = __expf(a0 - mnew);
      const float p1 = __expf(a1 - mnew);
      float sum = p0 + p1;
#pragma unroll
      for (int off = 8; off >= 1; off >>= 1) sum += __shfl_xor(sum, off, 32);
      lrow[r] = lrow[r] * alpha + sum;
      mrow[r] = mnew;
#pragma unroll
      for (int j = 0; j < 4; ++j) acc[j][r] *= alpha;
      const int m = r + half * 8;
      ld_p[wave][m][ln]      = f2bf(p0);
      ld_p[wave][m][16 + ln] = f2bf(p1);
    }

    // P (16x32) A-fragment back from per-wave LDS scratch; acc += P x V
    const unsigned short* pp = &ld_p[wave][ln][0];
    const int ab = half ? 16 : 0;
    v16bf pa = load_frag(pp + ab, pp + ab + 8);
#pragma unroll
    for (int j = 0; j < 4; ++j) {
      const unsigned short* pv = &ld_v[j * 16 + ln][half * 16];
      v16bf vb = load_frag(pv, pv + 8);
      acc[j] = WMMA_BF16(pa, vb, acc[j]);
    }
  }

  const int b = bh >> 3, h = bh & 7;
#pragma unroll
  for (int j = 0; j < 4; ++j) {
    const int cc = h * 64 + j * 16 + ln;
#pragma unroll
    for (int r = 0; r < 8; ++r) {
      const int s = qt * 64 + wave * 16 + r + half * 8;
      O[((size_t)b * 1024 + s) * 512 + cc] = f2bf(acc[j][r] / lrow[r]);
    }
  }
}

// ---------------------------------------------------------------- out-proj + bias + residual (f32 out)
// Computes out^T tile: D[c, row] = sum_k wproj[k,c] * O[row,k]  (A = W^T, B = O^T)
// => C/D lanes stripe the s dimension: stores & residual loads fully coalesced.
// WG tile: 128 c x 64 rows; wave tile 32 c x 64 rows (8 WMMA / k-chunk).
__global__ __launch_bounds__(128) void proj_kernel(const unsigned short* __restrict__ O,
                                                   const unsigned short* __restrict__ wp,
                                                   const float* __restrict__ bp,
                                                   const float* __restrict__ x,
                                                   float* __restrict__ out) {
  __shared__ unsigned short ld_w[128][32];  // [c_local][k_local]  (A operand)
  __shared__ unsigned short ld_o[64][32];   // [row_local][k_local] (B operand)
  const int c0 = blockIdx.x * 128;
  const int r0 = blockIdx.y * 64;           // global row = b*1024 + s
  const int t    = threadIdx.x;
  const int wave = t >> 5, lane = t & 31, half = lane >> 4, ln = lane & 15;

  const v8f zero8 = {0.f, 0.f, 0.f, 0.f, 0.f, 0.f, 0.f, 0.f};
  v8f acc[2][4] = {{zero8, zero8, zero8, zero8}, {zero8, zero8, zero8, zero8}};

  for (int k0 = 0; k0 < 512; k0 += 32) {
    __syncthreads();
    {
      int rr = t >> 1, kb = (t & 1) * 16;
      const unsigned short* ap = O + (size_t)(r0 + rr) * 512 + k0 + kb;
      *(uint4*)&ld_o[rr][kb]     = *(const uint4*)ap;
      *(uint4*)&ld_o[rr][kb + 8] = *(const uint4*)(ap + 8);
      int kk = t >> 2, cb = (t & 3) * 32;
      const unsigned short* wpp = wp + (size_t)(k0 + kk) * 512 + c0 + cb;
      U16x16 wv;
#pragma unroll
      for (int hw = 0; hw < 2; ++hw) {
        wv.u[0] = *(const uint4*)(wpp + hw * 16);
        wv.u[1] = *(const uint4*)(wpp + hw * 16 + 8);
#pragma unroll
        for (int i = 0; i < 16; ++i) ld_w[cb + hw * 16 + i][kk] = wv.s[i];
      }
    }
    __syncthreads();

    const int ab = half ? 8 : 0;
    const unsigned short* pa0 = &ld_w[wave * 32 + ln][0];
    const unsigned short* pa1 = &ld_w[wave * 32 + 16 + ln][0];
    v16bf a0 = load_frag(pa0 + ab, pa0 + ab + 16);
    v16bf a1 = load_frag(pa1 + ab, pa1 + ab + 16);
#pragma unroll
    for (int j = 0; j < 4; ++j) {
      const unsigned short* pb = &ld_o[j * 16 + ln][0];
      v16bf bb = load_frag(pb + half * 8, pb + half * 8 + 16);
      acc[0][j] = WMMA_BF16(a0, bb, acc[0][j]);
      acc[1][j] = WMMA_BF16(a1, bb, acc[1][j]);
    }
  }

#pragma unroll
  for (int m2 = 0; m2 < 2; ++m2) {
#pragma unroll
    for (int r = 0; r < 8; ++r) {
      const int cc = c0 + wave * 32 + m2 * 16 + r + half * 8;   // lane-uniform
      const float bias = bp[cc];
#pragma unroll
      for (int j = 0; j < 4; ++j) {
        const int row = r0 + j * 16 + ln;                        // lanes stripe s
        const int b = row >> 10, s = row & 1023;
        const size_t idx = ((size_t)b * 512 + cc) * 1024 + s;
        out[idx] = acc[m2][j][r] + bias + x[idx];
      }
    }
  }
}

// ---------------------------------------------------------------- launch
extern "C" void kernel_launch(void* const* d_in, const int* in_sizes, int n_in,
                              void* d_out, int out_size, void* d_ws, size_t ws_size,
                              hipStream_t stream) {
  const float* x      = (const float*)d_in[0];
  const float* w_qkv  = (const float*)d_in[1];
  const float* b_qkv  = (const float*)d_in[2];
  const float* w_proj = (const float*)d_in[3];
  const float* b_proj = (const float*)d_in[4];
  float* out = (float*)d_out;

  char* ws = (char*)d_ws;
  unsigned short* wq_bf = (unsigned short*)(ws + 0);          // 512*1536*2 = 1,572,864
  unsigned short* wp_bf = (unsigned short*)(ws + 1572864);    // 512*512*2  =   524,288
  unsigned short* Qb    = (unsigned short*)(ws + 2097152);    // 16 MB each [B,H,S,64] bf16
  unsigned short* Kb    = (unsigned short*)(ws + 18874368);
  unsigned short* Vb    = (unsigned short*)(ws + 35651584);
  unsigned short* Ob    = (unsigned short*)(ws + 52428800);   // [B*S, C] bf16, ends ~66 MB

  prep_kernel<<<3072, 256, 0, stream>>>(w_qkv, w_proj, wq_bf, wp_bf);
  qkv_kernel<<<dim3(24, 8, 16), 128, 0, stream>>>(x, wq_bf, b_qkv, Qb, Kb, Vb);
  attn_kernel<<<dim3(16, 128), 128, 0, stream>>>(Qb, Kb, Vb, Ob);
  proj_kernel<<<dim3(4, 256), 128, 0, stream>>>(Ob, wp_bf, b_proj, x, out);
}
